// GraphConvNet_64622077936093
// MI455X (gfx1250) — matrix operand order repaired and checked by reading, not compiled, not run on hardware
//
#include <hip/hip_runtime.h>

#define N_NODES   50000
#define N_EDGES   800000
#define HID       128
#define OUT_CH    64
#define N_LAYERS  3
#define NUM_GRAPHS 128
#define BN_EPS    1e-5f
#define KCHUNKS   4   // 128 / 32

typedef __attribute__((ext_vector_type(16))) __bf16 v16bf;
typedef __attribute__((ext_vector_type(8)))  float  v8f;

__device__ __forceinline__ v8f wmma_bf16(v16bf a, v16bf b, v8f c) {
  // D = A*B + C, f32 accumulate -> v_wmma_f32_16x16x32_bf16
  return __builtin_amdgcn_wmma_f32_16x16x32_bf16(
      false, a, false, b, (short)0, c, false, false);
}

// A fragment: 16x32 bf16 (MxK) from row-major f32, converted on load.
// lanes 0-15: M=lane,    elems 0-7 -> K=kc..+7,   8-15 -> K=kc+16..+23
// lanes16-31: M=lane-16, elems 0-7 -> K=kc+8..+15, 8-15 -> K=kc+24..+31
__device__ __forceinline__ v16bf load_a_frag(const float* __restrict__ A, int lda,
                                             int rowBase, int kc, int lane) {
  const int m  = rowBase + (lane & 15);
  const int kb = kc + ((lane & 16) ? 8 : 0);
  const float* p = A + (size_t)m * lda + kb;
  float4 q0 = *(const float4*)(p);
  float4 q1 = *(const float4*)(p + 4);
  float4 q2 = *(const float4*)(p + 16);
  float4 q3 = *(const float4*)(p + 20);
  v16bf r;
  r[0]=(__bf16)q0.x; r[1]=(__bf16)q0.y; r[2]=(__bf16)q0.z; r[3]=(__bf16)q0.w;
  r[4]=(__bf16)q1.x; r[5]=(__bf16)q1.y; r[6]=(__bf16)q1.z; r[7]=(__bf16)q1.w;
  r[8]=(__bf16)q2.x; r[9]=(__bf16)q2.y; r[10]=(__bf16)q2.z; r[11]=(__bf16)q2.w;
  r[12]=(__bf16)q3.x; r[13]=(__bf16)q3.y; r[14]=(__bf16)q3.z; r[15]=(__bf16)q3.w;
  return r;
}

// Pre-swizzle a f32 weight [128 x N] into bf16 B fragments.
// Blob layout: fragment (colTile t, kChunk c) at ((t*KCHUNKS + c)*32 + lane)*16,
// lane's 16 contiguous bf16 = exactly its v16bf B operand.
// B frag semantics: N = t*16 + (lane&15), K = c*32 + ((lane&16)?8:0) + (e&7) + ((e&8)?16:0).
__global__ void __launch_bounds__(256) pack_w_kernel(
    const float* __restrict__ W, __bf16* __restrict__ out, int N) {
  const int tid  = blockIdx.x * 256 + threadIdx.x;
  const int e    = tid & 15;
  const int lane = (tid >> 4) & 31;
  const int ct   = tid >> 9;              // t*KCHUNKS + c
  const int c    = ct & (KCHUNKS - 1);
  const int t    = ct >> 2;
  const int kb   = c * 32 + ((lane & 16) ? 8 : 0) + (e & 7) + ((e & 8) ? 16 : 0);
  const int n    = t * 16 + (lane & 15);
  out[tid] = (__bf16)W[(size_t)kb * N + n];
}

__device__ __forceinline__ v16bf load_b_packed(const __bf16* __restrict__ Wp,
                                               int chunk, int colTile, int lane) {
  const __bf16* p = Wp + ((((size_t)colTile * KCHUNKS + chunk) * 32 + lane) << 4);
  return *(const v16bf*)p;   // 32B contiguous -> 2x global_load_b128
}

// h0 = x @ W + b ; 16 rows/block, 8 waves = 8 column tiles (128 cols)
__global__ void __launch_bounds__(256) gemm_init_kernel(
    const float* __restrict__ x, const __bf16* __restrict__ Wp,
    const float* __restrict__ bias, float* __restrict__ out) {
  const int wave = threadIdx.x >> 5, lane = threadIdx.x & 31;
  const int rowBase = blockIdx.x * 16;
  v8f c = {};
#pragma unroll
  for (int kc = 0; kc < KCHUNKS; ++kc) {
    v16bf a = load_a_frag(x, HID, rowBase, kc * 32, lane);
    v16bf b = load_b_packed(Wp, kc, wave, lane);
    c = wmma_bf16(a, b, c);
  }
  const int col = wave * 16 + (lane & 15);
  const int rOff = (lane & 16) ? 8 : 0;
  const float bc = bias[col];
#pragma unroll
  for (int r = 0; r < 8; ++r)
    out[(size_t)(rowBase + rOff + r) * HID + col] = c[r] + bc;
}

// agg[dst] += h[src], one wave per edge, float4 per lane (L2-resident atomics)
__global__ void __launch_bounds__(256) scatter_kernel(
    const float* __restrict__ h, const int* __restrict__ src,
    const int* __restrict__ dst, float* __restrict__ agg) {
  const int t = blockIdx.x * 256 + threadIdx.x;
  const int edge = t >> 5, lane = t & 31;
  if (edge >= N_EDGES) return;
  const int s = src[edge], d = dst[edge];
  const float4 v = ((const float4*)(h + (size_t)s * HID))[lane];
  float* a = agg + (size_t)d * HID + lane * 4;
  atomicAdd(a + 0, v.x); atomicAdd(a + 1, v.y);
  atomicAdd(a + 2, v.z); atomicAdd(a + 3, v.w);
}

// hnew = agg@relW + h@rootW + bias + h (in-place into agg) + BN stats
__global__ void __launch_bounds__(256) gemm_layer_kernel(
    const float* __restrict__ h, float* __restrict__ agg,
    const __bf16* __restrict__ relWp, const __bf16* __restrict__ rootWp,
    const float* __restrict__ bias, float* __restrict__ sum,
    float* __restrict__ sumsq) {
  const int wave = threadIdx.x >> 5, lane = threadIdx.x & 31;
  const int rowBase = blockIdx.x * 16;
  v8f c = {};
#pragma unroll
  for (int kc = 0; kc < KCHUNKS; ++kc) {
    v16bf aA = load_a_frag(agg, HID, rowBase, kc * 32, lane);
    v16bf bR = load_b_packed(relWp, kc, wave, lane);
    c = wmma_bf16(aA, bR, c);
    v16bf aH = load_a_frag(h, HID, rowBase, kc * 32, lane);
    v16bf bT = load_b_packed(rootWp, kc, wave, lane);
    c = wmma_bf16(aH, bT, c);
  }
  const int col = wave * 16 + (lane & 15);
  const int rOff = (lane & 16) ? 8 : 0;
  const float bc = bias[col];
  float vals[8], s = 0.f, s2 = 0.f;
#pragma unroll
  for (int r = 0; r < 8; ++r) {
    const size_t idx = (size_t)(rowBase + rOff + r) * HID + col;
    float v = c[r] + bc + h[idx];          // + residual
    vals[r] = v; s += v; s2 += v * v;
  }
  __syncthreads();  // all A-fragment reads of agg done before in-place store
#pragma unroll
  for (int r = 0; r < 8; ++r)
    agg[(size_t)(rowBase + rOff + r) * HID + col] = vals[r];
  s  += __shfl_xor(s, 16);
  s2 += __shfl_xor(s2, 16);
  if (lane < 16) { atomicAdd(&sum[col], s); atomicAdd(&sumsq[col], s2); }
}

__global__ void bn_params_kernel(const float* __restrict__ sum,
                                 const float* __restrict__ sumsq,
                                 const float* __restrict__ gamma,
                                 const float* __restrict__ beta,
                                 float* __restrict__ bnA, float* __restrict__ bnB) {
  const int c = threadIdx.x;
  const float inv_n = 1.0f / (float)N_NODES;
  const float mean = sum[c] * inv_n;
  const float var  = sumsq[c] * inv_n - mean * mean;
  const float a = gamma[c] * rsqrtf(var + BN_EPS);
  bnA[c] = a;
  bnB[c] = beta[c] - mean * a;
}

__global__ void __launch_bounds__(256) bn_apply_kernel(
    float* __restrict__ h, const float* __restrict__ bnA,
    const float* __restrict__ bnB) {
  const size_t i = (size_t)blockIdx.x * 256 + threadIdx.x;
  const int c = (int)(i & (HID - 1));
  const float v = h[i] * bnA[c] + bnB[c];
  h[i] = v > 0.f ? v : 0.f;
}

// out = agg@relW + h@rootW + b, pooled by graph id straight into d_out
__global__ void __launch_bounds__(256) gemm_final_kernel(
    const float* __restrict__ h, const float* __restrict__ agg,
    const __bf16* __restrict__ relWp, const __bf16* __restrict__ rootWp,
    const float* __restrict__ bias, const int* __restrict__ batch,
    float* __restrict__ pool) {
  const int wave = threadIdx.x >> 5, lane = threadIdx.x & 31;
  const int rowBase = blockIdx.x * 32 + (wave >> 2) * 16;
  const int colTile = wave & 3;
  if (rowBase >= N_NODES) return;  // wave-uniform: EXEC all-ones for WMMA
  v8f c = {};
#pragma unroll
  for (int kc = 0; kc < KCHUNKS; ++kc) {
    v16bf aA = load_a_frag(agg, HID, rowBase, kc * 32, lane);
    v16bf bR = load_b_packed(relWp, kc, colTile, lane);
    c = wmma_bf16(aA, bR, c);
    v16bf aH = load_a_frag(h, HID, rowBase, kc * 32, lane);
    v16bf bT = load_b_packed(rootWp, kc, colTile, lane);
    c = wmma_bf16(aH, bT, c);
  }
  const int col = colTile * 16 + (lane & 15);
  const int rOff = (lane & 16) ? 8 : 0;
  const float bc = bias[col];
#pragma unroll
  for (int r = 0; r < 8; ++r) {
    const int row = rowBase + rOff + r;
    const int g = batch[row];
    atomicAdd(&pool[(size_t)g * OUT_CH + col], c[r] + bc);
  }
}

extern "C" void kernel_launch(void* const* d_in, const int* in_sizes, int n_in,
                              void* d_out, int out_size, void* d_ws, size_t ws_size,
                              hipStream_t stream) {
  const float* x       = (const float*)d_in[0];
  const int*   eidx    = (const int*)d_in[1];
  const int*   batch   = (const int*)d_in[2];
  const float* lin_w   = (const float*)d_in[3];
  const float* lin_b   = (const float*)d_in[4];
  const float* rel_w   = (const float*)d_in[5];
  const float* root_w  = (const float*)d_in[6];
  const float* conv_b  = (const float*)d_in[7];
  const float* gamma   = (const float*)d_in[8];
  const float* beta    = (const float*)d_in[9];
  const float* rel_wl  = (const float*)d_in[10];
  const float* root_wl = (const float*)d_in[11];
  const float* b_last  = (const float*)d_in[12];

  const int* src = eidx;
  const int* dst = eidx + N_EDGES;

  float* buf0  = (float*)d_ws;
  float* buf1  = buf0 + (size_t)N_NODES * HID;
  float* sum   = buf1 + (size_t)N_NODES * HID;
  float* sumsq = sum + HID;
  float* bnA   = sumsq + HID;
  float* bnB   = bnA + HID;
  __bf16* wpBase = (__bf16*)(bnB + HID);   // 32B-aligned (all offsets mult. of 32)

  const size_t WSZ = (size_t)HID * HID;    // 16384 elems
  const size_t LSZ = (size_t)HID * OUT_CH; // 8192 elems
  __bf16* wp_lin    = wpBase;
  __bf16* wp_rel0   = wp_lin + WSZ;        // rel 0..2, then root 0..2
  __bf16* wp_root0  = wp_rel0 + 3 * WSZ;
  __bf16* wp_rel_l  = wp_root0 + 3 * WSZ;
  __bf16* wp_root_l = wp_rel_l + LSZ;

  const size_t nodeBytes = (size_t)N_NODES * HID * sizeof(float);
  const int gemmGrid = N_NODES / 16;          // 3125, exact
  const int scatGrid = (N_EDGES * 32) / 256;  // 100000, exact
  const int bnGrid   = (N_NODES * HID) / 256; // 25000, exact

  // One-time weight swizzle f32 -> bf16 WMMA B-fragment layout
  pack_w_kernel<<<WSZ / 256, 256, 0, stream>>>(lin_w, wp_lin, HID);
  for (int i = 0; i < N_LAYERS; ++i) {
    pack_w_kernel<<<WSZ / 256, 256, 0, stream>>>(rel_w  + (size_t)i * WSZ, wp_rel0  + (size_t)i * WSZ, HID);
    pack_w_kernel<<<WSZ / 256, 256, 0, stream>>>(root_w + (size_t)i * WSZ, wp_root0 + (size_t)i * WSZ, HID);
  }
  pack_w_kernel<<<LSZ / 256, 256, 0, stream>>>(rel_wl,  wp_rel_l,  OUT_CH);
  pack_w_kernel<<<LSZ / 256, 256, 0, stream>>>(root_wl, wp_root_l, OUT_CH);

  gemm_init_kernel<<<gemmGrid, 256, 0, stream>>>(x, wp_lin, lin_b, buf0);

  float* hbuf = buf0;
  float* abuf = buf1;
  for (int i = 0; i < N_LAYERS; ++i) {
    hipMemsetAsync(abuf, 0, nodeBytes, stream);
    hipMemsetAsync(sum, 0, 2 * HID * sizeof(float), stream);
    scatter_kernel<<<scatGrid, 256, 0, stream>>>(hbuf, src, dst, abuf);
    gemm_layer_kernel<<<gemmGrid, 256, 0, stream>>>(
        hbuf, abuf, wp_rel0 + (size_t)i * WSZ, wp_root0 + (size_t)i * WSZ,
        conv_b + (size_t)i * HID, sum, sumsq);
    bn_params_kernel<<<1, HID, 0, stream>>>(sum, sumsq, gamma + (size_t)i * HID,
                                            beta + (size_t)i * HID, bnA, bnB);
    bn_apply_kernel<<<bnGrid, 256, 0, stream>>>(abuf, bnA, bnB);
    float* t = hbuf; hbuf = abuf; abuf = t;
  }

  hipMemsetAsync(abuf, 0, nodeBytes, stream);
  scatter_kernel<<<scatGrid, 256, 0, stream>>>(hbuf, src, dst, abuf);
  hipMemsetAsync(d_out, 0, (size_t)NUM_GRAPHS * OUT_CH * sizeof(float), stream);
  gemm_final_kernel<<<(N_NODES + 31) / 32, 256, 0, stream>>>(
      hbuf, abuf, wp_rel_l, wp_root_l, b_last, batch, (float*)d_out);
}
